// Mamba_73143293050998
// MI455X (gfx1250) — matrix operand order repaired and checked by reading, not compiled, not run on hardware
//
#include <hip/hip_runtime.h>
#include <stdint.h>

#define V_SZ    50257
#define D_MODEL 768
#define NLAYER  4
#define D_INNER 1536
#define D_STATE 16
#define D_CONV  4
#define BATCH   2
#define SEQ     512
#define BL      (BATCH * SEQ)   // 1024 rows of activation

typedef __attribute__((ext_vector_type(16))) __bf16 v16bf;
typedef __attribute__((ext_vector_type(8)))  float  v8f;
typedef __attribute__((address_space(3))) uint32_t lds_u32;

union Frag { v16bf v; uint32_t u[8]; };

__device__ __forceinline__ uint32_t f2bf_u(float f) {
  uint32_t x = __float_as_uint(f);
  return (x + 0x7FFFu + ((x >> 16) & 1u)) >> 16;   // round-to-nearest-even bf16
}
__device__ __forceinline__ uint32_t pack_bf2(float lo, float hi) {
  return f2bf_u(lo) | (f2bf_u(hi) << 16);
}
__device__ __forceinline__ float silu_f(float v) {
  return v / (1.0f + __expf(-v));
}

// -------- CDNA5 async global->LDS staging (bypasses VGPRs, ASYNCcnt) --------
__device__ __forceinline__ void async_ld_b128(uint32_t* lds, const uint32_t* gsrc) {
  asm volatile("global_load_async_to_lds_b128 %0, %1, off"
               :: "v"((lds_u32*)lds), "v"(gsrc)
               : "memory");
}
__device__ __forceinline__ void wait_async_le4() {
  asm volatile("s_wait_asynccnt 0x4" ::: "memory");
}
__device__ __forceinline__ void wait_async_0() {
  asm volatile("s_wait_asynccnt 0x0" ::: "memory");
}

// ---------------- f32 -> packed bf16 conversion ----------------
__global__ __launch_bounds__(256) void k_cvt_bf16(const float* __restrict__ in,
                                                  uint32_t* __restrict__ out,
                                                  int npairs) {
  int i = blockIdx.x * 256 + threadIdx.x;
  if (i < npairs) out[i] = pack_bf2(in[2 * i], in[2 * i + 1]);
}

// ---------------- embedding gather (f32 + bf16 copies) ----------------
__global__ __launch_bounds__(256) void k_embed(const int* __restrict__ idx,
                                               const float* __restrict__ emb,
                                               float* __restrict__ x,
                                               uint32_t* __restrict__ x_bf) {
  int m = blockIdx.x;
  const float* src = emb + (size_t)idx[m] * D_MODEL;
  float* dst = x + (size_t)m * D_MODEL;
  for (int d = threadIdx.x; d < D_MODEL; d += 256) dst[d] = src[d];
  for (int p = threadIdx.x; p < D_MODEL / 2; p += 256)
    x_bf[(size_t)m * (D_MODEL / 2) + p] = pack_bf2(src[2 * p], src[2 * p + 1]);
}

// ---------------- WMMA GEMM: C[M][N] (f32) = A[M][K] * B[N][K]^T (bf16) ----
// Block tile 128x128, 8 waves, each wave owns a 32x64 slab (2x4 WMMA frags).
// Double-buffered LDS fed by global_load_async_to_lds_b128.
#define TM 128
#define TN 128
#define TK 32
#define LDSW 17   // dwords (=34 bf16) per LDS row; gcd(17,64)==1 -> no bank conflicts

__global__ __launch_bounds__(256) void k_gemm_bf16(
    const uint32_t* __restrict__ A,    // bf16 pairs, row-major [M][K/2]
    const uint32_t* __restrict__ Bm,   // bf16 pairs, row-major [N][K/2]
    float* __restrict__ C,             // [M][N]
    const float* __restrict__ Cin,     // optional residual (may alias C, may be null)
    int M, int N, int K) {
  __shared__ uint32_t sA[2][TM * LDSW];
  __shared__ uint32_t sB[2][TN * LDSW];
  const int tid   = threadIdx.x;
  const int lane  = tid & 31;
  const int wave  = tid >> 5;
  const int waveM = wave & 3;   // 4 waves along M (32 rows each)
  const int waveN = wave >> 2;  // 2 waves along N (64 cols each)
  const int row   = lane & 15;
  const int hi    = lane >> 4;  // lane half
  const int bm = blockIdx.y * TM;
  const int bn = blockIdx.x * TN;
  const int Kd = K >> 1;

  __builtin_amdgcn_s_wait_tensorcnt(0);   // no TDM outstanding (CDNA5 marker, no-op)

  // per-thread staging geometry (constant across tiles):
  // tiles are 128 rows x 16 dwords; 256 threads x 2 segments x 16B each.
  const int r0 = tid >> 2;               // 0..63
  const int cd = (tid & 3) * 4;          // dword column within tile
  // B rows beyond N only feed output columns that the store guard discards,
  // so clamp the address instead of zero-filling (stays in-bounds).
  const int nb0 = (bn + r0      < N) ? (bn + r0)      : (N - 1);
  const int nb1 = (bn + r0 + 64 < N) ? (bn + r0 + 64) : (N - 1);

  auto issue_tile = [&](int buf, int kd0) {
    async_ld_b128(&sA[buf][r0 * LDSW + cd],        A  + (size_t)(bm + r0)      * Kd + kd0 + cd);
    async_ld_b128(&sA[buf][(r0 + 64) * LDSW + cd], A  + (size_t)(bm + r0 + 64) * Kd + kd0 + cd);
    async_ld_b128(&sB[buf][r0 * LDSW + cd],        Bm + (size_t)nb0            * Kd + kd0 + cd);
    async_ld_b128(&sB[buf][(r0 + 64) * LDSW + cd], Bm + (size_t)nb1            * Kd + kd0 + cd);
  };

  v8f acc[2][4] = {};
  const int nk = K / TK;

  issue_tile(0, 0);
  for (int t = 0; t < nk; ++t) {
    if (t + 1 < nk) {
      issue_tile((t + 1) & 1, ((t + 1) * TK) >> 1);
      wait_async_le4();                 // tile t's 4 async loads complete (in-order)
    } else {
      wait_async_0();
    }
    __syncthreads();                    // tile t visible workgroup-wide

    const uint32_t* bufA = sA[t & 1];
    const uint32_t* bufB = sB[t & 1];
    Frag fa[2], fb[4];
#pragma unroll
    for (int i = 0; i < 2; ++i) {
      int rb = waveM * 32 + i * 16 + row;
#pragma unroll
      for (int v = 0; v < 8; ++v) {
        // ISA A layout: VGPR v holds K pair at ((v&4)?16:0) + hi*8 + 2*(v&3)
        int kbd = ((v & 4) ? 8 : 0) + hi * 4 + (v & 3);
        fa[i].u[v] = bufA[rb * LDSW + kbd];
      }
    }
#pragma unroll
    for (int j = 0; j < 4; ++j) {
      int rb = waveN * 64 + j * 16 + row;
#pragma unroll
      for (int v = 0; v < 8; ++v) {
        // ISA B layout: VGPR v holds K pair at ((v&4)?16:0) + hi*8 + 2*(v&3)
        int kbd = ((v & 4) ? 8 : 0) + hi * 4 + (v & 3);
        fb[j].u[v] = bufB[rb * LDSW + kbd];
      }
    }
#pragma unroll
    for (int i = 0; i < 2; ++i)
#pragma unroll
      for (int j = 0; j < 4; ++j)
        acc[i][j] = __builtin_amdgcn_wmma_f32_16x16x32_bf16(
            false, fa[i].v, false, fb[j].v, (short)0, acc[i][j], false, false);
    __syncthreads();                    // all waves done reading buf t before reuse
  }

  // store: C/D layout -> VGPR v holds M = hi*8 + v, lane(row) is N
#pragma unroll
  for (int i = 0; i < 2; ++i) {
#pragma unroll
    for (int j = 0; j < 4; ++j) {
      int ng = bn + waveN * 64 + j * 16 + row;
      if (ng < N) {
#pragma unroll
        for (int v = 0; v < 8; ++v) {
          int mg = bm + waveM * 32 + i * 16 + hi * 8 + v;
          size_t off = (size_t)mg * N + ng;
          float o = acc[i][j][v];
          if (Cin) o += Cin[off];
          C[off] = o;
        }
      }
    }
  }
}

// ---------------- depthwise causal conv(4) + SiLU ----------------
__global__ __launch_bounds__(256) void k_conv_silu(
    const float* __restrict__ xz,        // [BL][2*DI], xs half = [:, :DI]
    const float* __restrict__ cw,        // [DI][DC] (layer slice)
    float* __restrict__ xs,              // [BL][DI]
    uint16_t* __restrict__ xs_bf) {
  int c = blockIdx.x * 256 + threadIdx.x;  // channel
  int m = blockIdx.y;                      // row (b*SEQ + l)
  int l = m & (SEQ - 1);
  float w0 = cw[c * D_CONV + 0], w1 = cw[c * D_CONV + 1];
  float w2 = cw[c * D_CONV + 2], w3 = cw[c * D_CONV + 3];
  float acc = w3 * xz[(size_t)m * (2 * D_INNER) + c];
  if (l >= 1) acc += w2 * xz[(size_t)(m - 1) * (2 * D_INNER) + c];
  if (l >= 2) acc += w1 * xz[(size_t)(m - 2) * (2 * D_INNER) + c];
  if (l >= 3) acc += w0 * xz[(size_t)(m - 3) * (2 * D_INNER) + c];
  float s = silu_f(acc);
  xs[(size_t)m * D_INNER + c] = s;
  xs_bf[(size_t)m * D_INNER + c] = (uint16_t)f2bf_u(s);
}

// ---------------- xp = xs @ W_xproj^T (N=33, tiny) ----------------
__global__ __launch_bounds__(256) void k_xproj(const float* __restrict__ xs,
                                               const float* __restrict__ Wxp, // [33][DI]
                                               float* __restrict__ xp) {      // [BL][33]
  int m = blockIdx.x;
  __shared__ float srow[D_INNER];
  for (int k = threadIdx.x; k < D_INNER; k += 256)
    srow[k] = xs[(size_t)m * D_INNER + k];
  __syncthreads();
  for (int e = threadIdx.x; e < 2 * D_STATE + 1; e += 256) {
    const float* wr = Wxp + (size_t)e * D_INNER;
    float acc = 0.f;
    for (int k = 0; k < D_INNER; ++k) acc += srow[k] * wr[k];
    xp[(size_t)m * (2 * D_STATE + 1) + e] = acc;
  }
}

// ---------------- selective scan (one thread per (batch, channel)) --------
__global__ __launch_bounds__(256) void k_scan(
    const float* __restrict__ xp,     // [BL][33]: dt_raw, B[16], C[16]
    const float* __restrict__ xs,     // [BL][DI]
    const float* __restrict__ xz,     // [BL][2*DI], z half = [:, DI:]
    const float* __restrict__ W_dt,   // [DI]
    const float* __restrict__ b_dt,   // [DI]
    const float* __restrict__ a_log,  // [DI][DS]
    const float* __restrict__ d_par,  // [DI]
    uint16_t* __restrict__ y_bf) {    // [BL][DI]
  int t = blockIdx.x * 256 + threadIdx.x;
  if (t >= BATCH * D_INNER) return;
  int b = t / D_INNER;
  int c = t - b * D_INNER;

  float A[D_STATE], h[D_STATE];
#pragma unroll
  for (int n = 0; n < D_STATE; ++n) {
    A[n] = -__expf(a_log[(size_t)c * D_STATE + n]);
    h[n] = 0.f;
  }
  float wdt = W_dt[c], bdt = b_dt[c], dpar = d_par[c];

  for (int l = 0; l < SEQ; ++l) {
    int m = b * SEQ + l;
    const float* xpr = xp + (size_t)m * (2 * D_STATE + 1);
    float draw = xpr[0] * wdt + bdt;
    float dlt  = (draw > 20.f) ? draw : log1pf(__expf(draw));  // softplus
    float xv = xs[(size_t)m * D_INNER + c];
    float zv = xz[(size_t)m * (2 * D_INNER) + D_INNER + c];
    float dx = dlt * xv;
    float accv = 0.f;
#pragma unroll
    for (int n = 0; n < D_STATE; ++n) {
      float abar = __expf(dlt * A[n]);
      h[n] = abar * h[n] + dx * xpr[1 + n];
      accv += xpr[1 + D_STATE + n] * h[n];
    }
    float yv = (accv + dpar * xv) * silu_f(zv);
    y_bf[(size_t)m * D_INNER + c] = (uint16_t)f2bf_u(yv);
  }
}

// ---------------- RMSNorm -> packed bf16 ----------------
__global__ __launch_bounds__(256) void k_rmsnorm(const float* __restrict__ x,
                                                 const float* __restrict__ nw,
                                                 uint32_t* __restrict__ xn_bf) {
  int m = blockIdx.x;
  __shared__ float red[256];
  float s = 0.f;
  for (int d = threadIdx.x; d < D_MODEL; d += 256) {
    float v = x[(size_t)m * D_MODEL + d];
    s += v * v;
  }
  red[threadIdx.x] = s;
  __syncthreads();
  for (int o = 128; o > 0; o >>= 1) {
    if (threadIdx.x < o) red[threadIdx.x] += red[threadIdx.x + o];
    __syncthreads();
  }
  float rms = rsqrtf(red[0] / (float)D_MODEL + 1e-5f);
  for (int p = threadIdx.x; p < D_MODEL / 2; p += 256) {
    float a = x[(size_t)m * D_MODEL + 2 * p]     * rms * nw[2 * p];
    float b = x[(size_t)m * D_MODEL + 2 * p + 1] * rms * nw[2 * p + 1];
    xn_bf[(size_t)m * (D_MODEL / 2) + p] = pack_bf2(a, b);
  }
}

extern "C" void kernel_launch(void* const* d_in, const int* in_sizes, int n_in,
                              void* d_out, int out_size, void* d_ws, size_t ws_size,
                              hipStream_t stream) {
  (void)in_sizes; (void)n_in; (void)out_size; (void)ws_size;
  const int*   idx     = (const int*)  d_in[0];
  const float* emb     = (const float*)d_in[1];
  const float* W_in    = (const float*)d_in[2];
  const float* conv_w  = (const float*)d_in[3];
  const float* W_xproj = (const float*)d_in[4];
  const float* W_dt    = (const float*)d_in[5];
  const float* b_dt    = (const float*)d_in[6];
  const float* a_log   = (const float*)d_in[7];
  const float* d_param = (const float*)d_in[8];
  const float* W_out   = (const float*)d_in[9];
  const float* norm_w  = (const float*)d_in[10];
  float* out = (float*)d_out;

  char* base = (char*)d_ws;
  size_t off = 0;
  auto alloc = [&](size_t bytes) -> void* {
    void* p = base + off;
    off = (off + bytes + 255) & ~(size_t)255;
    return p;
  };

  float*    x      = (float*)   alloc((size_t)BL * D_MODEL * 4);
  float*    xz     = (float*)   alloc((size_t)BL * 2 * D_INNER * 4);
  float*    xs     = (float*)   alloc((size_t)BL * D_INNER * 4);
  float*    xp     = (float*)   alloc((size_t)BL * (2 * D_STATE + 1) * 4);
  uint32_t* x_bf   = (uint32_t*)alloc((size_t)BL * D_MODEL * 2);
  uint32_t* xn_bf  = (uint32_t*)alloc((size_t)BL * D_MODEL * 2);
  uint16_t* xs_bf  = (uint16_t*)alloc((size_t)BL * D_INNER * 2);
  uint16_t* y_bf   = (uint16_t*)alloc((size_t)BL * D_INNER * 2);
  uint32_t* emb_bf = (uint32_t*)alloc((size_t)V_SZ * D_MODEL * 2);
  uint32_t* Win_bf = (uint32_t*)alloc((size_t)NLAYER * 2 * D_INNER * D_MODEL * 2);
  uint32_t* Wout_bf= (uint32_t*)alloc((size_t)NLAYER * D_MODEL * D_INNER * 2);

  // one-time bf16 conversions (embedding doubles as the logit weight matrix)
  {
    int np = V_SZ * D_MODEL / 2;
    k_cvt_bf16<<<dim3((np + 255) / 256), dim3(256), 0, stream>>>(emb, emb_bf, np);
    np = NLAYER * 2 * D_INNER * D_MODEL / 2;
    k_cvt_bf16<<<dim3((np + 255) / 256), dim3(256), 0, stream>>>(W_in, Win_bf, np);
    np = NLAYER * D_MODEL * D_INNER / 2;
    k_cvt_bf16<<<dim3((np + 255) / 256), dim3(256), 0, stream>>>(W_out, Wout_bf, np);
  }

  k_embed<<<dim3(BL), dim3(256), 0, stream>>>(idx, emb, x, x_bf);

  for (int l = 0; l < NLAYER; ++l) {
    const uint32_t* Wi = Win_bf + (size_t)l * 2 * D_INNER * D_MODEL / 2;
    const uint32_t* Wo = Wout_bf + (size_t)l * D_MODEL * D_INNER / 2;

    // xz = x @ W_in^T   (1024 x 3072 x 768)
    k_gemm_bf16<<<dim3((2 * D_INNER + TN - 1) / TN, BL / TM), dim3(256), 0, stream>>>(
        x_bf, Wi, xz, nullptr, BL, 2 * D_INNER, D_MODEL);

    k_conv_silu<<<dim3(D_INNER / 256, BL), dim3(256), 0, stream>>>(
        xz, conv_w + (size_t)l * D_INNER * D_CONV, xs, xs_bf);

    k_xproj<<<dim3(BL), dim3(256), 0, stream>>>(
        xs, W_xproj + (size_t)l * (2 * D_STATE + 1) * D_INNER, xp);

    k_scan<<<dim3((BATCH * D_INNER + 255) / 256), dim3(256), 0, stream>>>(
        xp, xs, xz,
        W_dt + (size_t)l * D_INNER, b_dt + (size_t)l * D_INNER,
        a_log + (size_t)l * D_INNER * D_STATE, d_param + (size_t)l * D_INNER, y_bf);

    // x += y @ W_out^T  (1024 x 768 x 1536), residual fused at store
    k_gemm_bf16<<<dim3((D_MODEL + TN - 1) / TN, BL / TM), dim3(256), 0, stream>>>(
        (const uint32_t*)y_bf, Wo, x, x, BL, D_MODEL, D_INNER);

    // refresh bf16 residual stream for next layer's A matrix
    k_cvt_bf16<<<dim3((BL * D_MODEL / 2 + 255) / 256), dim3(256), 0, stream>>>(
        x, x_bf, BL * D_MODEL / 2);
  }

  k_rmsnorm<<<dim3(BL), dim3(256), 0, stream>>>(x, norm_w, xn_bf);

  // logits = xn @ emb^T  (1024 x 50257 x 768) — emb tile lives in the 192MB L2
  k_gemm_bf16<<<dim3((V_SZ + TN - 1) / TN, BL / TM), dim3(256), 0, stream>>>(
      xn_bf, emb_bf, out, nullptr, BL, V_SZ, D_MODEL);
}
